// MSPathCatSampler_24816321036792
// MI455X (gfx1250) — matrix operand order repaired and checked by reading, not compiled, not run on hardware
//
#include <hip/hip_runtime.h>
#include <math.h>

// Problem constants (B, L, V, H, R) = (256, 512, 64, 256, 8)
constexpr int Bv   = 256;
constexpr int Lc   = 512;
constexpr int Vc   = 64;
constexpr int Hc   = 256;
constexpr int MAXR = 15;          // 2*R-1
constexpr int Dc   = Lc * Vc;     // 32768

typedef __attribute__((ext_vector_type(16))) __bf16 v16bf;
typedef __attribute__((ext_vector_type(8)))  __bf16 v8bf;
typedef __attribute__((ext_vector_type(4)))  __bf16 v4bf;
typedef __attribute__((ext_vector_type(8)))  float  v8f;
typedef __attribute__((ext_vector_type(4)))  float  v4f;

// ---------------- deterministic hash RNG (stand-in for threefry) -----------
__device__ __forceinline__ unsigned hash_u32(unsigned x) {
    x ^= x >> 16; x *= 0x7feb352dU;
    x ^= x >> 15; x *= 0x846ca68bU;
    x ^= x >> 16;
    return x;
}
__device__ __forceinline__ float u01(unsigned h) {
    return (float)(h >> 8) * (1.0f / 16777216.0f) + 1.0e-7f;
}
__device__ __forceinline__ float gumbel(unsigned h) {
    float u = u01(h);
    return -__logf(-__logf(u));
}

// ---------------- 1) one-hot -> integer labels -----------------------------
__global__ void k_labels(const float* __restrict__ x, int* __restrict__ lab) {
    int i = blockIdx.x * blockDim.x + threadIdx.x;   // over B*L
    if (i >= Bv * Lc) return;
    const float* xr = x + (size_t)i * Vc;
    int m = 0;
    #pragma unroll 8
    for (int v = 0; v < Vc; ++v) if (xr[v] > 0.5f) m = v;
    lab[i] = m;
}

// ---------------- 2) W1 f32 -> bf16 (GEMM operand) -------------------------
__global__ void k_w1bf(const float* __restrict__ W, __bf16* __restrict__ Wb) {
    size_t i = (size_t)blockIdx.x * blockDim.x + threadIdx.x; // over D*H/4
    if (i >= (size_t)Dc * Hc / 4) return;
    v4f w = ((const v4f*)W)[i];
    v4bf o;
    o[0] = (__bf16)w[0]; o[1] = (__bf16)w[1];
    o[2] = (__bf16)w[2]; o[3] = (__bf16)w[3];
    ((v4bf*)Wb)[i] = o;
}

// ---------------- 3) h = tanh(gather-sum), a = (1-h^2)*w2, score -----------
__global__ void k_h(const float* __restrict__ W1, const float* __restrict__ w2,
                    const int* __restrict__ labels,
                    __bf16* __restrict__ a_out, float* __restrict__ score) {
    int b = blockIdx.x, t = threadIdx.x;       // blockDim.x == Hc == 256
    __shared__ int   slab[Lc];
    __shared__ float red[256];
    for (int l = t; l < Lc; l += 256) slab[l] = labels[b * Lc + l];
    __syncthreads();
    float acc = 0.0f;
    for (int l = 0; l < Lc; ++l) {
        const float* row = W1 + ((size_t)(l * Vc + slab[l])) * Hc;
        if (l + 1 < Lc)
            __builtin_prefetch(W1 + ((size_t)((l + 1) * Vc + slab[l + 1])) * Hc + t, 0, 3);
        acc += row[t];
    }
    float th = tanhf(acc);
    a_out[b * Hc + t] = (__bf16)((1.0f - th * th) * w2[t]);
    red[t] = th * w2[t];
    __syncthreads();
    for (int s = 128; s > 0; s >>= 1) {
        if (t < s) red[t] += red[t + s];
        __syncthreads();
    }
    if (t == 0) score[b] = red[0];
}

// ---------------- 4) grad = a @ W1^T via bf16 WMMA -------------------------
// grid = (Dc/128, Bv/32), block = 256 (8 waves); each wave computes TWO 16x16
// output tiles sharing one B fragment (halves W1^T L2 traffic, 2 WMMA / B-load).
__global__ void k_gemm(const __bf16* __restrict__ A, const __bf16* __restrict__ Wb,
                       float* __restrict__ grad) {
    const int lane = threadIdx.x & 31;
    const int wave = threadIdx.x >> 5;
    const int b0 = blockIdx.y * 32;
    const int d0 = blockIdx.x * 128 + wave * 16;
    const int m  = lane & 15;       // A row within tile / B column within tile
    const int kg = lane >> 4;       // half-select per ISA 16-bit layouts

    const __bf16* arow0 = A  + (size_t)(b0 + m) * Hc;
    const __bf16* arow1 = arow0 + (size_t)16 * Hc;
    const __bf16* brow  = Wb + (size_t)(d0 + m) * Hc;

    v8f acc0 = {}, acc1 = {};
    #pragma unroll
    for (int k0 = 0; k0 < Hc; k0 += 32) {
        // A fragment (16x32 bf16): lanes<16 carry K {k0..k0+15},
        // lanes>=16 carry K {k0+16..k0+31}.
        v8bf a0lo = *(const v8bf*)(arow0 + k0 + kg * 16);
        v8bf a0hi = *(const v8bf*)(arow0 + k0 + 8 + kg * 16);
        v8bf a1lo = *(const v8bf*)(arow1 + k0 + kg * 16);
        v8bf a1hi = *(const v8bf*)(arow1 + k0 + 8 + kg * 16);
        v16bf af0, af1;
        #pragma unroll
        for (int e = 0; e < 8; ++e) {
            af0[e] = a0lo[e]; af0[8 + e] = a0hi[e];
            af1[e] = a1lo[e]; af1[8 + e] = a1hi[e];
        }
        // B fragment (32x16 bf16): lane gives N=m, K half = kg*16, 16 contiguous.
        v16bf bf = *(const v16bf*)(brow + k0 + kg * 16);
        if (k0 + 32 < Hc)
            __builtin_prefetch(brow + k0 + 32 + kg * 16, 0, 3);
        acc0 = __builtin_amdgcn_wmma_f32_16x16x32_bf16(
                   false, af0, false, bf, (short)0, acc0, false, false);
        acc1 = __builtin_amdgcn_wmma_f32_16x16x32_bf16(
                   false, af1, false, bf, (short)0, acc1, false, false);
    }
    // D/C layout: VGPR r -> M = r + 8*kg, N = m
    #pragma unroll
    for (int r = 0; r < 8; ++r) {
        grad[(size_t)(b0 + r + 8 * kg) * Dc + d0 + m]      = acc0[r];
        grad[(size_t)(b0 + 16 + r + 8 * kg) * Dc + d0 + m] = acc1[r];
    }
}

// ---------------- 5) lz[b,l] = logsumexp_v(grad/2) -------------------------
__global__ void k_zl(const float* __restrict__ grad, float* __restrict__ lz) {
    int i = blockIdx.x * blockDim.x + threadIdx.x;   // over B*L
    if (i >= Bv * Lc) return;
    const float* g = grad + (size_t)i * Vc;
    float mx = -1e30f;
    #pragma unroll 8
    for (int v = 0; v < Vc; ++v) mx = fmaxf(mx, 0.5f * g[v]);
    float s = 0.0f;
    #pragma unroll 8
    for (int v = 0; v < Vc; ++v) s += __expf(0.5f * g[v] - mx);
    lz[i] = mx + __logf(s);
}

// ---------------- 6) the 15-step path sampler (one block per batch row) ----
// 128KB grad row made LDS-resident via CDNA5 async memory->LDS DMA
// (GLOBAL_LOAD_ASYNC_TO_LDS_B128, tracked by ASYNCcnt; no VGPR round trip).
__global__ void k_scan(const float* __restrict__ grad, const float* __restrict__ lz,
                       const int* __restrict__ labx, const int* __restrict__ radius,
                       int* __restrict__ laby, int* __restrict__ idxlist,
                       float* __restrict__ logfwd) {
    const int b = blockIdx.x, t = threadIdx.x;
    extern __shared__ float smem[];
    float* gRow = smem;                 // Dc floats (128KB)
    float* sl2  = gRow + Dc;            // Lc: lz[l] - S_l/2
    float* red  = sl2 + Lc;             // 256
    int*   redi = (int*)(red + 256);    // 256
    int*   lab  = redi + 256;           // Lc

    // --- async bulk fill: global grad row -> LDS, 16B granules ---
    {
        const float* gsrc = grad + (size_t)b * Dc;
        for (int d = t * 4; d < Dc; d += 256 * 4) {
            unsigned lds_addr = (unsigned)(size_t)(const void*)(gRow + d);
            unsigned long long ga = (unsigned long long)(size_t)(gsrc + d);
            asm volatile("global_load_async_to_lds_b128 %0, %1, off"
                         :: "v"(lds_addr), "v"(ga) : "memory");
        }
        asm volatile("s_wait_asynccnt 0" ::: "memory");
    }
    for (int l = t; l < Lc; l += 256) lab[l] = labx[b * Lc + l];
    __syncthreads();
    for (int l = t; l < Lc; l += 256)
        sl2[l] = lz[b * Lc + l] - 0.5f * gRow[(l << 6) + lab[l]];
    const int rad = radius[b];
    float lf = 0.0f;
    __syncthreads();

    for (int r = 0; r < MAXR; ++r) {
        // Gumbel-argmax categorical over D = 32768
        float best = -1e30f; int bidx = 0;
        const unsigned base = (unsigned)(r * Bv + b) * 32771u + 0x9e3779b9u;
        for (int d = t; d < Dc; d += 256) {
            int l = d >> 6;
            float sc = 0.5f * (gRow[d] - gRow[(l << 6) + lab[l]]);
            float v = sc + gumbel(hash_u32(base + (unsigned)d * 2654435761u));
            if (v > best) { best = v; bidx = d; }
        }
        red[t] = best; redi[t] = bidx;
        __syncthreads();
        for (int s = 128; s > 0; s >>= 1) {
            if (t < s && red[t + s] > red[t]) { red[t] = red[t + s]; redi[t] = redi[t + s]; }
            __syncthreads();
        }
        const int idx = redi[0];
        __syncthreads();

        // forward log-prob normalizer: logsumexp_l(sl2)
        float mx = -1e30f;
        for (int l = t; l < Lc; l += 256) mx = fmaxf(mx, sl2[l]);
        red[t] = mx; __syncthreads();
        for (int s = 128; s > 0; s >>= 1) {
            if (t < s) red[t] = fmaxf(red[t], red[t + s]);
            __syncthreads();
        }
        mx = red[0]; __syncthreads();
        float sm = 0.0f;
        for (int l = t; l < Lc; l += 256) sm += __expf(sl2[l] - mx);
        red[t] = sm; __syncthreads();
        for (int s = 128; s > 0; s >>= 1) {
            if (t < s) red[t] += red[t + s];
            __syncthreads();
        }
        if (t == 0) {
            const float logZ = mx + __logf(red[0]);
            idxlist[r * Bv + b] = idx;
            if (r < rad) {
                const int l = idx >> 6;
                const float chosen = 0.5f * (gRow[idx] - gRow[(l << 6) + lab[l]]);
                lf += chosen - logZ;
                lab[l] = idx & 63;
                sl2[l] = lz[b * Lc + l] - 0.5f * gRow[(l << 6) + lab[l]];
            }
        }
        __syncthreads();
    }
    for (int l = t; l < Lc; l += 256) laby[b * Lc + l] = lab[l];
    if (t == 0) logfwd[b] = lf;
}

// ---------------- 7) backward path log-prob (replay with grad_y) ----------
__global__ void k_logbwd(const float* __restrict__ grad, const float* __restrict__ lz,
                         const int* __restrict__ labx, const int* __restrict__ radius,
                         const int* __restrict__ idxlist, float* __restrict__ logbwd) {
    const int b = blockIdx.x, t = threadIdx.x;
    __shared__ float sl2[Lc];
    __shared__ float red[256];
    __shared__ int   lab[Lc];
    const float* g = grad + (size_t)b * Dc;
    for (int l = t; l < Lc; l += 256) {
        lab[l] = labx[b * Lc + l];
        sl2[l] = lz[b * Lc + l] - 0.5f * g[(l << 6) + lab[l]];
    }
    const int rad = radius[b];
    float acc = 0.0f;
    __syncthreads();
    for (int r = 0; r < MAXR; ++r) {
        // advance state to "after step r" (masked exactly like the scan)
        if (t == 0 && r < rad) {
            const int idx = idxlist[r * Bv + b];
            const int l = idx >> 6;
            lab[l] = idx & 63;
            sl2[l] = lz[b * Lc + l] - 0.5f * g[(l << 6) + lab[l]];
        }
        __syncthreads();
        float mx = -1e30f;
        for (int l = t; l < Lc; l += 256) mx = fmaxf(mx, sl2[l]);
        red[t] = mx; __syncthreads();
        for (int s = 128; s > 0; s >>= 1) {
            if (t < s) red[t] = fmaxf(red[t], red[t + s]);
            __syncthreads();
        }
        mx = red[0]; __syncthreads();
        float sm = 0.0f;
        for (int l = t; l < Lc; l += 256) sm += __expf(sl2[l] - mx);
        red[t] = sm; __syncthreads();
        for (int s = 128; s > 0; s >>= 1) {
            if (t < s) red[t] += red[t + s];
            __syncthreads();
        }
        if (t == 0) {
            const float logZ = mx + __logf(red[0]);
            const int idx = idxlist[r * Bv + b];
            const int l = idx >> 6;
            const float chosen = 0.5f * (g[idx] - g[(l << 6) + lab[l]]);
            acc += chosen - logZ;   // no mask on the backward sum
        }
        __syncthreads();
    }
    if (t == 0) logbwd[b] = acc;
}

// ---------------- 8) MH accept --------------------------------------------
__global__ void k_accept(const float* __restrict__ sx, const float* __restrict__ sy,
                         const float* __restrict__ lf, const float* __restrict__ lb,
                         int* __restrict__ accept) {
    int b = blockIdx.x * blockDim.x + threadIdx.x;
    if (b >= Bv) return;
    float la = (lb[b] + sy[b]) - (lf[b] + sx[b]);
    float u  = u01(hash_u32(0xC0FFEEu + (unsigned)b * 2654435761u));
    accept[b] = (__expf(la) >= u) ? 1 : 0;
}

// ---------------- 9) materialize output one-hot ----------------------------
__global__ void k_out(const int* __restrict__ labx, const int* __restrict__ laby,
                      const int* __restrict__ accept, float* __restrict__ out) {
    size_t i = (size_t)blockIdx.x * blockDim.x + threadIdx.x; // over B*L*V
    if (i >= (size_t)Bv * Lc * Vc) return;
    const int v  = (int)(i & (Vc - 1));
    const size_t bl = i >> 6;
    const int b  = (int)(bl >> 9);          // L = 512
    const int lab = accept[b] ? laby[bl] : labx[bl];
    out[i] = (v == lab) ? 1.0f : 0.0f;
}

// ===========================================================================
extern "C" void kernel_launch(void* const* d_in, const int* in_sizes, int n_in,
                              void* d_out, int out_size, void* d_ws, size_t ws_size,
                              hipStream_t stream) {
    (void)in_sizes; (void)n_in; (void)out_size; (void)ws_size;
    const float* x      = (const float*)d_in[0];   // (B,L,V)
    const float* W1     = (const float*)d_in[1];   // (D,H)
    const float* w2     = (const float*)d_in[2];   // (H)
    const int*   radius = (const int*)d_in[3];     // (B,1)

    char* p = (char*)d_ws;
    __bf16* w1b   = (__bf16*)p;  p += (size_t)Dc * Hc * sizeof(__bf16);   // 16 MB
    float* grad   = (float*)p;   p += (size_t)Bv * Dc * sizeof(float);    // 32 MB (x then y)
    __bf16* a_bf  = (__bf16*)p;  p += (size_t)Bv * Hc * sizeof(__bf16);
    int*   labx   = (int*)p;     p += (size_t)Bv * Lc * sizeof(int);
    int*   laby   = (int*)p;     p += (size_t)Bv * Lc * sizeof(int);
    float* lz     = (float*)p;   p += (size_t)Bv * Lc * sizeof(float);    // lz_x then lz_y
    int*   idxl   = (int*)p;     p += (size_t)MAXR * Bv * sizeof(int);
    float* scx    = (float*)p;   p += Bv * sizeof(float);
    float* scy    = (float*)p;   p += Bv * sizeof(float);
    float* lfwd   = (float*)p;   p += Bv * sizeof(float);
    float* lbwd   = (float*)p;   p += Bv * sizeof(float);
    int*   acc    = (int*)p;     p += Bv * sizeof(int);

    const dim3 blk(256);

    k_labels<<<(Bv * Lc + 255) / 256, blk, 0, stream>>>(x, labx);
    k_w1bf  <<<(Dc * Hc / 4 + 255) / 256, blk, 0, stream>>>(W1, w1b);

    // ---- forward state x ----
    k_h   <<<Bv, blk, 0, stream>>>(W1, w2, labx, a_bf, scx);
    k_gemm<<<dim3(Dc / 128, Bv / 32), blk, 0, stream>>>(a_bf, w1b, grad);
    k_zl  <<<(Bv * Lc + 255) / 256, blk, 0, stream>>>(grad, lz);

    const size_t scan_lds = (size_t)(Dc + Lc + 256) * sizeof(float)
                          + (size_t)(256 + Lc) * sizeof(int);   // ~137 KB
    k_scan<<<Bv, blk, scan_lds, stream>>>(grad, lz, labx, radius, laby, idxl, lfwd);

    // ---- proposed state y ----
    k_h   <<<Bv, blk, 0, stream>>>(W1, w2, laby, a_bf, scy);
    k_gemm<<<dim3(Dc / 128, Bv / 32), blk, 0, stream>>>(a_bf, w1b, grad);  // grad_y
    k_zl  <<<(Bv * Lc + 255) / 256, blk, 0, stream>>>(grad, lz);           // lz_y

    k_logbwd<<<Bv, blk, 0, stream>>>(grad, lz, labx, radius, idxl, lbwd);
    k_accept<<<1, 256, 0, stream>>>(scx, scy, lfwd, lbwd, acc);
    k_out   <<<(int)(((size_t)Bv * Lc * Vc + 255) / 256), blk, 0, stream>>>(
                labx, laby, acc, (float*)d_out);
}
